// UserModel_38663295598630
// MI455X (gfx1250) — compile-verified, matrix operand order checked by reading, not run on hardware
//
#include <hip/hip_runtime.h>
#include <hip/hip_bf16.h>
#include <math.h>

typedef __attribute__((ext_vector_type(16))) _Float16 v16h;
typedef __attribute__((ext_vector_type(8)))  _Float16 v8h;
typedef __attribute__((ext_vector_type(8)))  float    v8f;
typedef __attribute__((ext_vector_type(4)))  float    v4f;

#define DEV __device__ __forceinline__

constexpr int Bc  = 32;
constexpr int Lc  = 200;
constexpr int Vd  = 128;
constexpr int NCc = 256;
constexpr int NDc = 32;
constexpr int BL  = Bc * Lc;   // 6400

// ---------------------------------------------------------------------------
// WMMA 16x16x32 f16 helpers (wave32 layouts per CDNA5 ISA 7.12.2).
// Per lane the 16x32 fragment is two contiguous 8-half chunks:
//   row = lane&15, kb = (lane>>4)*8, halves 0-7 = K[kb..kb+8),
//   halves 8-15 = K[16+kb..16+kb+8)  ->  two b128 loads.
// ---------------------------------------------------------------------------
DEV v16h load_frag(const _Float16* p, int ld, int lane) {
  int r  = lane & 15;
  int kb = ((lane >> 4) & 1) << 3;
  const _Float16* q = p + r * ld + kb;
  v8h lo = *(const v8h*)(q);
  v8h hi = *(const v8h*)(q + 16);
  return __builtin_shufflevector(lo, hi, 0, 1, 2, 3, 4, 5, 6, 7,
                                 8, 9, 10, 11, 12, 13, 14, 15);
}

DEV v8f wmma16(v16h a, v16h b, v8f c) {
  return __builtin_amdgcn_wmma_f32_16x16x32_f16(
      false, a, false, b, (short)0, c, false, false);
}

// ---------------------------------------------------------------------------
// Small prep kernels
// ---------------------------------------------------------------------------
__global__ void k_zero_f32(float* p, int n) {
  int i = blockIdx.x * blockDim.x + threadIdx.x;
  if (i < n) p[i] = 0.0f;
}

__global__ void k_cvt_f16(const float* __restrict__ s, _Float16* __restrict__ d, int n) {
  int i = blockIdx.x * blockDim.x + threadIdx.x;
  if (i < n) d[i] = (_Float16)s[i];
}

// W2a is [128 x 384]; extract columns 128..383 as f16 row-major [128 x 256]
__global__ void k_extract_w2ag(const float* __restrict__ W2a, _Float16* __restrict__ d) {
  int i = blockIdx.x * blockDim.x + threadIdx.x;  // 0..32767
  int j = i >> 8, k = i & 255;
  d[i] = (_Float16)W2a[j * 384 + 128 + k];
}

// u[j] = sum_k v_beta[k] * W2a[j][k],  k in [0,128)
__global__ void k_compute_u(const float* __restrict__ W2a,
                            const float* __restrict__ v_beta,
                            float* __restrict__ u) {
  int j = threadIdx.x;  // 128 threads
  float s = 0.f;
  for (int k = 0; k < 128; ++k) s += v_beta[k] * W2a[j * 384 + k];
  u[j] = s;
}

// gin_h[bl][0:128] = f16(X[c+256*d]);  gin_h[bl][128:256] = f16(r * v_r)
__global__ void k_embed(const int* __restrict__ c_seq, const int* __restrict__ d_seq,
                        const float* __restrict__ r_seq, const float* __restrict__ X,
                        const float* __restrict__ v_r, _Float16* __restrict__ gin) {
  int bl = blockIdx.x;       // 0..6399 (row = b*L + l)
  int v  = threadIdx.x;      // 0..127
  int idx = c_seq[bl] + NCc * d_seq[bl];
  gin[(size_t)bl * 256 + v]       = (_Float16)X[(size_t)idx * Vd + v];
  gin[(size_t)bl * 256 + 128 + v] = (_Float16)(r_seq[bl] * v_r[v]);
}

// ---------------------------------------------------------------------------
// Parallel WMMA GEMM:  out[M x N] = A[M x K](f16) @ W[N x K]^T + bias
// One wave computes a 16 x (16*NT) strip: A fragment loaded once per k-step,
// NT accumulators chained back-to-back through the WMMA pipe.
// ---------------------------------------------------------------------------
template <int NT, int RELU, int OUTF16>
__global__ void k_wmma_gemm(const _Float16* __restrict__ A,
                            const _Float16* __restrict__ W,
                            const float* __restrict__ bias,
                            void* __restrict__ Out,
                            int M, int N, int K) {
  int lane = threadIdx.x & 31;
  int wid  = (blockIdx.x * blockDim.x + threadIdx.x) >> 5;
  int ngrp  = (N >> 4) / NT;
  int total = (M >> 4) * ngrp;
  if (wid >= total) return;
  int mt = wid / ngrp, ng = wid % ngrp;

  const _Float16* Ap = A + (size_t)(mt * 16) * K;
  int n  = lane & 15;
  int mb = ((lane >> 4) & 1) * 8;

  v8f acc[NT];
#pragma unroll
  for (int i = 0; i < NT; ++i) {
    float bv = bias[(ng * NT + i) * 16 + n];
#pragma unroll
    for (int r = 0; r < 8; ++r) acc[i][r] = bv;
  }

  for (int k = 0; k < K; k += 32) {
    v16h a = load_frag(Ap + k, K, lane);
#pragma unroll
    for (int i = 0; i < NT; ++i) {
      const _Float16* Wp = W + (size_t)((ng * NT + i) * 16) * K;
      v16h b = load_frag(Wp + k, K, lane);
      acc[i] = wmma16(a, b, acc[i]);
    }
  }

#pragma unroll
  for (int i = 0; i < NT; ++i)
#pragma unroll
    for (int r = 0; r < 8; ++r) {
      float val = acc[i][r];
      if (RELU) val = fmaxf(val, 0.0f);
      size_t o = (size_t)(mt * 16 + mb + r) * N + (ng * NT + i) * 16 + n;
      if (OUTF16) ((_Float16*)Out)[o] = (_Float16)val;
      else        ((float*)Out)[o]    = val;
    }
}

// ---------------------------------------------------------------------------
// GRU scan: one workgroup, 512 threads (16 waves), 200 sequential steps.
// W_hh is staged once into LDS (96 KB); per step each wave does 8+24
// ds_load_b128 and 12 chained WMMAs, then fused gate math.
// LDS total: 48K (gh) + 8K (h16) + 96K (whh_s) = 152 KB (< WGP LDS).
// ---------------------------------------------------------------------------
__global__ void __launch_bounds__(512, 1)
k_gru_scan(const float* __restrict__ gx,
           const _Float16* __restrict__ Whh,
           const float* __restrict__ b_hh,
           float* __restrict__ h_out,        // [B,L,128] f32
           _Float16* __restrict__ h_h) {     // [B,L,128] f16
  __shared__ __align__(16) float    gh[32 * 384];       // 48 KB
  __shared__ __align__(16) _Float16 h16[32 * 128];      //  8 KB
  __shared__ __align__(16) _Float16 whh_s[384 * 128];   // 96 KB
  int tid = threadIdx.x, lane = tid & 31, wid = tid >> 5;

  float hreg[8];
#pragma unroll
  for (int e = 0; e < 8; ++e) hreg[e] = 0.0f;
  for (int i = tid; i < 32 * 128; i += 512) h16[i] = (_Float16)0.0f;
  // stage W_hh (f16) into LDS: 6144 x 16B chunks
  for (int i = tid; i < (384 * 128) / 8; i += 512)
    ((v8h*)whh_s)[i] = ((const v8h*)Whh)[i];

  int mt = wid & 1;
  int ng = wid >> 1;  // 0..7 -> 3 N-tiles each (24 total)
  int n  = lane & 15;
  int mb = ((lane >> 4) & 1) * 8;

  float bhv[3];
#pragma unroll
  for (int i = 0; i < 3; ++i) bhv[i] = b_hh[(ng * 3 + i) * 16 + n];

  // gate-phase element -> gx base for prefetch
  const int eb = tid >> 7;          // first element's batch (e=0: idx=tid)
  const int ev = tid & 127;

  for (int t = 0; t < Lc; ++t) {
    __syncthreads();  // h16 + whh_s ready
    // prefetch this step's gx operands for the gate phase
    {
      const float* g = gx + ((size_t)eb * Lc + t) * 384;
      __builtin_prefetch(g + ev, 0, 0);
      __builtin_prefetch(g + 128 + ev, 0, 0);
      __builtin_prefetch(g + 256 + ev, 0, 0);
    }
    v16h af[4];
#pragma unroll
    for (int kk = 0; kk < 4; ++kk)
      af[kk] = load_frag(h16 + mt * 16 * 128 + kk * 32, 128, lane);  // LDS
#pragma unroll
    for (int i = 0; i < 3; ++i) {
      int nt = ng * 3 + i;
      v8f acc;
#pragma unroll
      for (int r = 0; r < 8; ++r) acc[r] = bhv[i];
#pragma unroll
      for (int kk = 0; kk < 4; ++kk) {
        v16h b = load_frag(whh_s + nt * 16 * 128 + kk * 32, 128, lane);  // LDS
        acc = wmma16(af[kk], b, acc);
      }
#pragma unroll
      for (int r = 0; r < 8; ++r)
        gh[(mt * 16 + mb + r) * 384 + nt * 16 + n] = acc[r];
    }
    __syncthreads();  // gh ready
#pragma unroll
    for (int e = 0; e < 8; ++e) {
      int idx = tid + e * 512;          // 0..4095
      int b = idx >> 7, v = idx & 127;
      size_t row = (size_t)b * Lc + t;
      const float* g = gx + row * 384;
      float xr = g[v], xz = g[128 + v], xn = g[256 + v];
      float hr = gh[b * 384 + v], hz = gh[b * 384 + 128 + v], hn = gh[b * 384 + 256 + v];
      float rg = 1.0f / (1.0f + expf(-(xr + hr)));
      float zg = 1.0f / (1.0f + expf(-(xz + hz)));
      float nn = tanhf(xn + rg * hn);
      float hv = (1.0f - zg) * nn + zg * hreg[e];
      hreg[e] = hv;
      h_out[row * 128 + v] = hv;
      h_h[row * 128 + v]   = (_Float16)hv;
      h16[idx]             = (_Float16)hv;
    }
  }
}

// ---------------------------------------------------------------------------
// Memory scan: one workgroup, 128 threads (4 waves), 200 sequential steps.
// W2b fragments hoisted into VGPRs (only 4/wave -> no spill). Per step:
// scalar gather, 32x128 relu, 4 WMMAs, row scatter into running C.
// ---------------------------------------------------------------------------
__global__ void __launch_bounds__(128, 1)
k_mem_scan(const int* __restrict__ c_seq, const int* __restrict__ d_seq,
           const float* __restrict__ pre2, const float* __restrict__ u,
           const _Float16* __restrict__ W2b, const float* __restrict__ b2b,
           float* __restrict__ C_run, float* __restrict__ newc) {
  __shared__ __align__(16) _Float16 act[32 * 128];
  __shared__ float beta_s[32];
  int tid = threadIdx.x, lane = tid & 31, wid = tid >> 5;
  int mt = wid & 1, nt = wid >> 1;   // 2x2 tiles over 32x32 output
  int n  = lane & 15;
  int mb = ((lane >> 4) & 1) * 8;

  // loop-invariant weight fragments + bias
  v16h wfrag[4];
#pragma unroll
  for (int kk = 0; kk < 4; ++kk)
    wfrag[kk] = load_frag(W2b + (size_t)(nt * 16) * 128 + kk * 32, 128, lane);
  float bv = b2b[nt * 16 + n];
  float uv = u[tid];

  for (int t = 0; t < Lc; ++t) {
    if (tid < 32) {
      int b = tid;
      int cc = c_seq[b * Lc + t], dd = d_seq[b * Lc + t];
      beta_s[b] = C_run[((size_t)b * NCc + cc) * NDc + dd];
    }
    __syncthreads();
    for (int b = 0; b < 32; ++b) {
      float a = pre2[((size_t)(b * Lc + t)) * 128 + tid] + beta_s[b] * uv;
      act[b * 128 + tid] = (_Float16)fmaxf(a, 0.0f);
    }
    __syncthreads();
    {
      v8f acc;
#pragma unroll
      for (int r = 0; r < 8; ++r) acc[r] = bv;
#pragma unroll
      for (int kk = 0; kk < 4; ++kk) {
        v16h a = load_frag(act + mt * 16 * 128 + kk * 32, 128, lane);  // LDS
        acc = wmma16(a, wfrag[kk], acc);
      }
#pragma unroll
      for (int r = 0; r < 8; ++r) {
        int bb = mt * 16 + mb + r;            // batch index
        int nn = nt * 16 + n;                 // d index
        float val = acc[r];
        newc[((size_t)(bb * Lc + t)) * NDc + nn] = val;
        int cc = c_seq[bb * Lc + t];
        C_run[((size_t)bb * NCc + cc) * NDc + nn] = val;
      }
    }
    __threadfence_block();   // C_run stores visible before next step's gather
    __syncthreads();
  }
}

// ---------------------------------------------------------------------------
// C_seq reconstruction (the 210 MB stream): one wave per (b, 4-c-group),
// lane = (c_sub, d/4). Forward fill over t; 512 B contiguous nontemporal
// b128 stores per wave per step.
// ---------------------------------------------------------------------------
__global__ void k_cseq_fill(const int* __restrict__ c_seq,
                            const float* __restrict__ newc,
                            float* __restrict__ Cseq) {
  int wid  = (blockIdx.x * blockDim.x + threadIdx.x) >> 5;  // 0..2047
  int lane = threadIdx.x & 31;
  int b   = wid >> 6;                // 0..31
  int cg  = wid & 63;                // 0..63
  int c   = cg * 4 + (lane >> 3);    // this lane's c row
  int dof = (lane & 7) * 4;          // 4 d-values per lane
  const int* crow = c_seq + b * Lc;
  v4f cur = {0.f, 0.f, 0.f, 0.f};
  for (int t = 0; t < Lc; ++t) {
    int ct = crow[t];
    size_t bl = (size_t)(b * Lc + t);
    if (ct == c) cur = *(const v4f*)(newc + bl * NDc + dof);
    __builtin_nontemporal_store(cur, (v4f*)(Cseq + (bl * NCc + c) * NDc + dof));
  }
}

// ---------------------------------------------------------------------------
// Host launcher
// ---------------------------------------------------------------------------
extern "C" void kernel_launch(void* const* d_in, const int* in_sizes, int n_in,
                              void* d_out, int out_size, void* d_ws, size_t ws_size,
                              hipStream_t stream) {
  const int*   c_seq  = (const int*)d_in[0];
  const int*   d_seq  = (const int*)d_in[1];
  const float* r_seq  = (const float*)d_in[2];
  const float* X      = (const float*)d_in[3];
  const float* v_r    = (const float*)d_in[4];
  const float* v_beta = (const float*)d_in[5];
  const float* W_ih   = (const float*)d_in[6];
  const float* W_hh   = (const float*)d_in[7];
  const float* b_ih   = (const float*)d_in[8];
  const float* b_hh   = (const float*)d_in[9];
  const float* W1a    = (const float*)d_in[10];
  const float* b1a    = (const float*)d_in[11];
  const float* W1b    = (const float*)d_in[12];
  const float* b1b    = (const float*)d_in[13];
  const float* W2a    = (const float*)d_in[14];
  const float* b2a    = (const float*)d_in[15];
  const float* W2b    = (const float*)d_in[16];
  const float* b2b    = (const float*)d_in[17];

  float* alpha_out = (float*)d_out;                       // [B,L,32]
  float* h_out     = alpha_out + (size_t)BL * NDc;        // [B,L,128]
  float* Cseq_out  = alpha_out + 1024000;                 // [B,L,256,32]

  // workspace carve-out (256B aligned slabs)
  size_t off = 0;
  auto take = [&](size_t bytes) {
    void* p = (char*)d_ws + off;
    off += (bytes + 255) & ~(size_t)255;
    return p;
  };
  _Float16* gin_h  = (_Float16*)take((size_t)BL * 256 * 2);
  _Float16* Wih_h  = (_Float16*)take(384 * 256 * 2);
  _Float16* Whh_h  = (_Float16*)take(384 * 128 * 2);
  _Float16* W1a_h  = (_Float16*)take(128 * 128 * 2);
  _Float16* W1b_h  = (_Float16*)take(32 * 128 * 2);
  _Float16* W2b_h  = (_Float16*)take(32 * 128 * 2);
  _Float16* W2ag_h = (_Float16*)take(128 * 256 * 2);
  float*    u128   = (float*)take(128 * 4);
  float*    gx     = (float*)take((size_t)BL * 384 * 4);
  float*    pre2   = (float*)take((size_t)BL * 128 * 4);
  _Float16* h_h    = (_Float16*)take((size_t)BL * 128 * 2);
  _Float16* tmp1_h = (_Float16*)take((size_t)BL * 128 * 2);
  float*    newc   = (float*)take((size_t)BL * NDc * 4);
  float*    C_run  = (float*)take((size_t)Bc * NCc * NDc * 4);

  // 0) zero running memory
  k_zero_f32<<<(Bc * NCc * NDc + 255) / 256, 256, 0, stream>>>(C_run, Bc * NCc * NDc);

  // 1) weight preparation (f32 -> f16), derived tensors
  k_cvt_f16<<<(384 * 256 + 255) / 256, 256, 0, stream>>>(W_ih, Wih_h, 384 * 256);
  k_cvt_f16<<<(384 * 128 + 255) / 256, 256, 0, stream>>>(W_hh, Whh_h, 384 * 128);
  k_cvt_f16<<<(128 * 128 + 255) / 256, 256, 0, stream>>>(W1a, W1a_h, 128 * 128);
  k_cvt_f16<<<(32 * 128 + 255) / 256, 256, 0, stream>>>(W1b, W1b_h, 32 * 128);
  k_cvt_f16<<<(32 * 128 + 255) / 256, 256, 0, stream>>>(W2b, W2b_h, 32 * 128);
  k_extract_w2ag<<<128, 256, 0, stream>>>(W2a, W2ag_h);
  k_compute_u<<<1, 128, 0, stream>>>(W2a, v_beta, u128);

  // 2) embedding + gru_in (f16)
  k_embed<<<BL, 128, 0, stream>>>(c_seq, d_seq, r_seq, X, v_r, gin_h);

  // 3) gx = gru_in @ W_ih^T + b_ih   (6400x256x384): 400 x 6 wave-strips
  k_wmma_gemm<4, 0, 0><<<300, 256, 0, stream>>>(gin_h, Wih_h, b_ih, gx,
                                                BL, 384, 256);
  // 4) pre2 = gru_in @ W2a[:,128:]^T + b2a  (6400x256x128): 400 x 2 strips
  k_wmma_gemm<4, 0, 0><<<100, 256, 0, stream>>>(gin_h, W2ag_h, b2a, pre2,
                                                BL, 128, 256);

  // 5) sequential GRU scan -> h_seq (f32 to d_out, f16 mirror for linear_1)
  k_gru_scan<<<1, 512, 0, stream>>>(gx, Whh_h, b_hh, h_out, h_h);

  // 6) linear_1: relu(h @ W1a^T + b1a) -> f16, then @ W1b^T + b1b -> alpha
  k_wmma_gemm<4, 1, 1><<<100, 256, 0, stream>>>(h_h, W1a_h, b1a, tmp1_h,
                                                BL, 128, 128);
  k_wmma_gemm<2, 0, 0><<<50, 256, 0, stream>>>(tmp1_h, W1b_h, b1b, alpha_out,
                                               BL, 32, 128);

  // 7) sequential memory scan -> newc (+ running C)
  k_mem_scan<<<1, 128, 0, stream>>>(c_seq, d_seq, pre2, u128, W2b_h, b2b,
                                    C_run, newc);

  // 8) parallel forward-fill of the 210MB C_seq snapshot stream
  k_cseq_fill<<<256, 256, 0, stream>>>(c_seq, newc, Cseq_out);
}
